// min_max_net_61555471286542
// MI455X (gfx1250) — compile-verified, matrix operand order checked by reading, not run on hardware
//
#include <hip/hip_runtime.h>
#include <hip/hip_bf16.h>
#include <math.h>

#define N_GROUPC   64
#define GROUP_SZ   64
#define INPUT_DIM  2048
#define BATCH      8192
#define N_MONTN    512
#define OUT_DIM    (N_GROUPC * GROUP_SZ)   // 4096

// ---- WMMA vector types (CDNA5 gfx1250, wave32) ----
typedef __attribute__((ext_vector_type(16))) __bf16 v16bf;
typedef __attribute__((ext_vector_type(8)))  float  v8f;

// ---------------- helpers ----------------
__device__ __forceinline__ unsigned short f32_to_bf16_rne(float v) {
    unsigned u = __float_as_uint(v);
    unsigned r = u + 0x7FFFu + ((u >> 16) & 1u);
    return (unsigned short)(r >> 16);
}

__device__ __forceinline__ void split2bf16(float v, unsigned short& h, unsigned short& l) {
    unsigned short hs = f32_to_bf16_rne(v);
    float fh = __uint_as_float(((unsigned)hs) << 16);
    h = hs;
    l = f32_to_bf16_rne(v - fh);
}

// CDNA5 async global->LDS copy (ASYNCcnt-tracked, no VGPR data path).
// LDS destination VGPR = low 32 bits of the flat shared-aperture address,
// which per the ISA aperture mapping is the LDS byte offset.
__device__ __forceinline__ void async_ld_b128(unsigned lds_off, const void* gaddr) {
    asm volatile("global_load_async_to_lds_b128 %0, %1, off"
                 :: "v"(lds_off), "v"(gaddr)
                 : "memory");
}

__device__ __forceinline__ void wait_async0() {
    asm volatile("s_wait_asynccnt 0" ::: "memory");
}

// ---------------- prep kernels ----------------
__global__ void k_col_flags(const int* __restrict__ montn, int* __restrict__ flags) {
    int j = blockIdx.x * blockDim.x + threadIdx.x;
    if (j >= INPUT_DIM) return;
    int f = 0;
    for (int k = 0; k < N_MONTN; ++k) f |= (montn[k] == j) ? 1 : 0;
    flags[j] = f;
}

__global__ void k_init_gmax(unsigned* __restrict__ gmax) {
    int i = blockIdx.x * blockDim.x + threadIdx.x;
    if (i < BATCH) gmax[i] = 0u;   // below monotone-encoding of every finite float
}

// split x (f32 -> bf16 hi/lo), 4 elems / thread
__global__ void k_split_x(const float* __restrict__ x,
                          unsigned short* __restrict__ xh,
                          unsigned short* __restrict__ xl) {
    size_t t = (size_t)blockIdx.x * blockDim.x + threadIdx.x;
    size_t base = t * 4;
    float4 v = *(const float4*)(x + base);
    unsigned short h0,h1,h2,h3,l0,l1,l2,l3;
    split2bf16(v.x, h0, l0); split2bf16(v.y, h1, l1);
    split2bf16(v.z, h2, l2); split2bf16(v.w, h3, l3);
    uint2 ph, pl;
    ph.x = (unsigned)h0 | ((unsigned)h1 << 16);
    ph.y = (unsigned)h2 | ((unsigned)h3 << 16);
    pl.x = (unsigned)l0 | ((unsigned)l1 << 16);
    pl.y = (unsigned)l2 | ((unsigned)l3 << 16);
    *(uint2*)(xh + base) = ph;
    *(uint2*)(xl + base) = pl;
}

// split w with monotone-column squaring applied, 4 elems / thread
__global__ void k_split_w(const float* __restrict__ w,
                          const int* __restrict__ flags,
                          unsigned short* __restrict__ wh,
                          unsigned short* __restrict__ wl) {
    size_t t = (size_t)blockIdx.x * blockDim.x + threadIdx.x;
    size_t base = t * 4;
    int col = (int)(base & (size_t)(INPUT_DIM - 1));  // 4 consecutive stay in-row
    float4 v = *(const float4*)(w + base);
    float f0 = flags[col + 0] ? v.x * v.x : v.x;
    float f1 = flags[col + 1] ? v.y * v.y : v.y;
    float f2 = flags[col + 2] ? v.z * v.z : v.z;
    float f3 = flags[col + 3] ? v.w * v.w : v.w;
    unsigned short h0,h1,h2,h3,l0,l1,l2,l3;
    split2bf16(f0, h0, l0); split2bf16(f1, h1, l1);
    split2bf16(f2, h2, l2); split2bf16(f3, h3, l3);
    uint2 ph, pl;
    ph.x = (unsigned)h0 | ((unsigned)h1 << 16);
    ph.y = (unsigned)h2 | ((unsigned)h3 << 16);
    pl.x = (unsigned)l0 | ((unsigned)l1 << 16);
    pl.y = (unsigned)l2 | ((unsigned)l3 << 16);
    *(uint2*)(wh + base) = ph;
    *(uint2*)(wl + base) = pl;
}

// ---------------- fused GEMM + group-min + global max ----------------
#define MB   64    // rows per workgroup (2 waves in M, 2 tiles each)
#define NBLK 128   // cols per workgroup = 2 groups
#define KC   32    // K chunk (bf16 WMMA K)
#define STR  40    // padded LDS row stride in bf16 elems (80 B)

__global__ __launch_bounds__(256)
void k_gemm_groupmin(const unsigned short* __restrict__ xh,
                     const unsigned short* __restrict__ xl,
                     const unsigned short* __restrict__ wh,
                     const unsigned short* __restrict__ wl,
                     const float* __restrict__ bias,
                     unsigned* __restrict__ gmax) {
    __shared__ unsigned short sAh[2][MB * STR];
    __shared__ unsigned short sAl[2][MB * STR];
    __shared__ unsigned short sBh[2][NBLK * STR];
    __shared__ unsigned short sBl[2][NBLK * STR];
    __shared__ float sPart[8][MB];

    const int tid  = threadIdx.x;
    const int lane = tid & 31;
    const int wid  = tid >> 5;     // 8 waves
    const int mh   = wid >> 2;     // 0..1 : wave's 32-row slice
    const int np   = wid & 3;      // 0..3 : wave's 32-col slice
    const int m0   = blockIdx.y * MB;
    const int n0   = blockIdx.x * NBLK;

    // cooperative async-staging coords (256 threads)
    const int ar = tid >> 2, ac = (tid & 3) * 8;    // A: 64 rows x 32 K, 1 x b128/array
    const int br = tid >> 1, bc = (tid & 1) * 16;   // B: 128 rows x 32 K, 2 x b128/array

    // WMMA 16-bit A/B fragment layout: lanes 0-15 row fr, K {0..7,16..23};
    // lanes 16-31 same rows, K {8..15,24..31}
    const int fr = lane & 15;
    const int kh = lane >> 4;
    int aoff[2], boff[2];
    aoff[0] = (mh * 32 +  0 + fr) * STR + kh * 8;
    aoff[1] = (mh * 32 + 16 + fr) * STR + kh * 8;
    boff[0] = (np * 32 +  0 + fr) * STR + kh * 8;
    boff[1] = (np * 32 + 16 + fr) * STR + kh * 8;

    v8f acc[2][2];
#pragma unroll
    for (int i = 0; i < 2; ++i)
#pragma unroll
        for (int j = 0; j < 2; ++j)
            acc[i][j] = (v8f){0.f,0.f,0.f,0.f,0.f,0.f,0.f,0.f};

    // per-thread global staging bases
    const unsigned short* gAh = xh + (size_t)(m0 + ar) * INPUT_DIM + ac;
    const unsigned short* gAl = xl + (size_t)(m0 + ar) * INPUT_DIM + ac;
    const unsigned short* gBh = wh + (size_t)(n0 + br) * INPUT_DIM + bc;
    const unsigned short* gBl = wl + (size_t)(n0 + br) * INPUT_DIM + bc;

    // per-buffer LDS destinations (low 32 bits of flat addr == LDS offset)
    unsigned dAh[2], dAl[2], dBh[2], dBl[2];
#pragma unroll
    for (int b = 0; b < 2; ++b) {
        dAh[b] = (unsigned)(uintptr_t)&sAh[b][ar * STR + ac];
        dAl[b] = (unsigned)(uintptr_t)&sAl[b][ar * STR + ac];
        dBh[b] = (unsigned)(uintptr_t)&sBh[b][br * STR + bc];
        dBl[b] = (unsigned)(uintptr_t)&sBl[b][br * STR + bc];
    }

    // prologue: async-fill buffer 0 with chunk 0
    {
        async_ld_b128(dAh[0], gAh);
        async_ld_b128(dAl[0], gAl);
        async_ld_b128(dBh[0],      gBh);
        async_ld_b128(dBh[0] + 16, gBh + 8);
        async_ld_b128(dBl[0],      gBl);
        async_ld_b128(dBl[0] + 16, gBl + 8);
    }

    for (int k0 = 0, it = 0; k0 < INPUT_DIM; k0 += KC, ++it) {
        const int cur = it & 1;
        wait_async0();        // my async loads into buf[cur] have landed
        __syncthreads();      // everyone's landed; buf[cur^1] fully consumed

        if (k0 + KC < INPUT_DIM) {       // async-prefetch next chunk
            const int nb = cur ^ 1;
            const int kn = k0 + KC;
            async_ld_b128(dAh[nb], gAh + kn);
            async_ld_b128(dAl[nb], gAl + kn);
            async_ld_b128(dBh[nb],      gBh + kn);
            async_ld_b128(dBh[nb] + 16, gBh + kn + 8);
            async_ld_b128(dBl[nb],      gBl + kn);
            async_ld_b128(dBl[nb] + 16, gBl + kn + 8);
        }

        union frag { v16bf v; uint4 q[2]; };
        frag fAh[2], fAl[2], fBh[2], fBl[2];
#pragma unroll
        for (int i = 0; i < 2; ++i) {
            fAh[i].q[0] = *(const uint4*)&sAh[cur][aoff[i]];
            fAh[i].q[1] = *(const uint4*)&sAh[cur][aoff[i] + 16];
            fAl[i].q[0] = *(const uint4*)&sAl[cur][aoff[i]];
            fAl[i].q[1] = *(const uint4*)&sAl[cur][aoff[i] + 16];
            fBh[i].q[0] = *(const uint4*)&sBh[cur][boff[i]];
            fBh[i].q[1] = *(const uint4*)&sBh[cur][boff[i] + 16];
            fBl[i].q[0] = *(const uint4*)&sBl[cur][boff[i]];
            fBl[i].q[1] = *(const uint4*)&sBl[cur][boff[i] + 16];
        }

        // split-precision fp32-via-bf16: xh*wh + xh*wl + xl*wh (12 WMMAs)
#pragma unroll
        for (int mt = 0; mt < 2; ++mt)
#pragma unroll
            for (int nt = 0; nt < 2; ++nt) {
                acc[mt][nt] = __builtin_amdgcn_wmma_f32_16x16x32_bf16(
                    false, fAh[mt].v, false, fBh[nt].v, (short)0, acc[mt][nt], false, false);
                acc[mt][nt] = __builtin_amdgcn_wmma_f32_16x16x32_bf16(
                    false, fAh[mt].v, false, fBl[nt].v, (short)0, acc[mt][nt], false, false);
                acc[mt][nt] = __builtin_amdgcn_wmma_f32_16x16x32_bf16(
                    false, fAl[mt].v, false, fBh[nt].v, (short)0, acc[mt][nt], false, false);
            }
    }

    // epilogue: + bias, per-row min over each tile's 16 cols (shfl within
    // 16-lane halves), stash per-(ntile,row) partials
#pragma unroll
    for (int mt = 0; mt < 2; ++mt)
#pragma unroll
        for (int nt = 0; nt < 2; ++nt) {
            float b = bias[n0 + np * 32 + nt * 16 + fr];
#pragma unroll
            for (int v = 0; v < 8; ++v) {
                float t = acc[mt][nt][v] + b;
                t = fminf(t, __shfl_xor(t, 1, 32));
                t = fminf(t, __shfl_xor(t, 2, 32));
                t = fminf(t, __shfl_xor(t, 4, 32));
                t = fminf(t, __shfl_xor(t, 8, 32));
                if (fr == 0)
                    sPart[np * 2 + nt][mh * 32 + mt * 16 + kh * 8 + v] = t;
            }
        }
    __syncthreads();

    if (tid < 2 * MB) {
        const int row = tid & (MB - 1);
        const int gl  = tid >> 6;          // 0/1: which of the 2 groups
        float g = fminf(fminf(sPart[gl * 4 + 0][row], sPart[gl * 4 + 1][row]),
                        fminf(sPart[gl * 4 + 2][row], sPart[gl * 4 + 3][row]));
        unsigned bits = __float_as_uint(g);
        unsigned enc  = (bits & 0x80000000u) ? ~bits : (bits | 0x80000000u);
        atomicMax(&gmax[m0 + row], enc);
    }
}

// ---------------- final sigmoid + Bernoulli ----------------
__global__ void k_sample(const unsigned* __restrict__ gmax,
                         const float* __restrict__ rand_u,
                         float* __restrict__ out) {
    int i = blockIdx.x * blockDim.x + threadIdx.x;
    if (i >= BATCH) return;
    unsigned e = gmax[i];
    unsigned bits = (e & 0x80000000u) ? (e ^ 0x80000000u) : ~e;
    float v = __uint_as_float(bits);
    float s = 1.0f / (1.0f + __expf(-v));
    out[i] = (rand_u[i] < s) ? 1.0f : 0.0f;
}

// ---------------- launch ----------------
extern "C" void kernel_launch(void* const* d_in, const int* in_sizes, int n_in,
                              void* d_out, int out_size, void* d_ws, size_t ws_size,
                              hipStream_t stream) {
    const float* x      = (const float*)d_in[0];
    const float* weight = (const float*)d_in[1];
    const float* bias   = (const float*)d_in[2];
    const float* rand_u = (const float*)d_in[3];
    const int*   montn  = (const int*)d_in[4];

    char* ws = (char*)d_ws;
    size_t off = 0;
    unsigned short* xh = (unsigned short*)(ws + off); off += (size_t)BATCH  * INPUT_DIM * 2;
    unsigned short* xl = (unsigned short*)(ws + off); off += (size_t)BATCH  * INPUT_DIM * 2;
    unsigned short* wh = (unsigned short*)(ws + off); off += (size_t)OUT_DIM * INPUT_DIM * 2;
    unsigned short* wl = (unsigned short*)(ws + off); off += (size_t)OUT_DIM * INPUT_DIM * 2;
    int*      flags = (int*)(ws + off);      off += (size_t)INPUT_DIM * 4;
    unsigned* gmax  = (unsigned*)(ws + off); off += (size_t)BATCH * 4;

    k_col_flags<<<(INPUT_DIM + 255) / 256, 256, 0, stream>>>(montn, flags);
    k_init_gmax<<<(BATCH + 255) / 256, 256, 0, stream>>>(gmax);
    k_split_x<<<(int)(((size_t)BATCH * INPUT_DIM / 4) / 256), 256, 0, stream>>>(x, xh, xl);
    k_split_w<<<(int)(((size_t)OUT_DIM * INPUT_DIM / 4) / 256), 256, 0, stream>>>(weight, flags, wh, wl);

    dim3 grid(N_GROUPC / 2, BATCH / MB);   // 32 group-pairs x 128 row-blocks
    k_gemm_groupmin<<<grid, 256, 0, stream>>>(xh, xl, wh, wl, bias, gmax);

    k_sample<<<(BATCH + 255) / 256, 256, 0, stream>>>(gmax, rand_u, (float*)d_out);
}
